// CorrFeatureFuser2D_72894184948155
// MI455X (gfx1250) — matrix operand order repaired and checked by reading, not compiled, not run on hardware
//
#include <hip/hip_runtime.h>

#define BATCH 4
#define NPTS  16384
#define HWPIX 65536
#define IMW   256
#define XSTR  104   // Xbuf row stride (halves)
#define FSTR  72    // F2buf row stride (halves)

typedef __attribute__((ext_vector_type(16))) _Float16 v16h;
typedef __attribute__((ext_vector_type(8)))  _Float16 v8h;
typedef __attribute__((ext_vector_type(2)))  _Float16 v2h;
typedef __attribute__((ext_vector_type(8)))  float    v8f;

static __device__ __forceinline__ unsigned pack2(float a, float b) {
  v2h v; v.x = (_Float16)a; v.y = (_Float16)b;
  return __builtin_bit_cast(unsigned, v);
}

static __device__ __forceinline__ v16h ld32h(const _Float16* p) {
  const v8h* q = (const v8h*)p;
  v8h lo = q[0], hi = q[1];
  return __builtin_shufflevector(lo, hi, 0,1,2,3,4,5,6,7,8,9,10,11,12,13,14,15);
}

// ---------------- Pre-pass 1: pack point data ----------------
// feat_3d [B,64,N] f32 -> ptF [B*N][64] f16 ; xy/flow3d -> ptS [B*N][4] f32
__global__ __launch_bounds__(256) void pack_points(
    const float* __restrict__ feat3d, const float* __restrict__ xy,
    const float* __restrict__ fl3, _Float16* __restrict__ ptF,
    float* __restrict__ ptS) {
  int t = blockIdx.x * 256 + threadIdx.x;          // 0 .. B*N-1
  int b = t >> 14, n = t & (NPTS - 1);
  const float* src = feat3d + (size_t)b * 64 * NPTS + n;
  unsigned u[32];
#pragma unroll
  for (int j = 0; j < 32; j++) {
    float a0 = src[(size_t)(2 * j) * NPTS];
    float a1 = src[(size_t)(2 * j + 1) * NPTS];
    u[j] = pack2(a0, a1);
  }
  uint4* dst = (uint4*)(ptF + ((size_t)t << 6));
#pragma unroll
  for (int j = 0; j < 8; j++)
    dst[j] = make_uint4(u[4*j], u[4*j+1], u[4*j+2], u[4*j+3]);
  const float* xyb = xy  + (size_t)b * 2 * NPTS;
  const float* flb = fl3 + (size_t)b * 2 * NPTS;
  float4 s;
  s.x = xyb[n]; s.y = xyb[NPTS + n];
  s.z = flb[n]; s.w = flb[NPTS + n];
  *(float4*)(ptS + 4 * (size_t)t) = s;
}

// ---------------- Pre-pass 2: weights -> CDNA5 A-fragment layout ----------------
// Fragment f(set,ktile,mtile): 32 lanes x 16 halves.
// A 16x32 frag as packed here (contiguous k, same as B): lane L: row M = mtile*16 + (L%16), g=L/16;
//   half i: k_local = g*16 + i
// Frag bases (frag units of 512 halves): W1(K=96):0..11, W2:12..19, W3:20..27, WF(K=128):28..43
__global__ __launch_bounds__(256) void pack_weights(
    const float* __restrict__ w1, const float* __restrict__ w2,
    const float* __restrict__ w3, const float* __restrict__ wf,
    _Float16* __restrict__ wsA) {
  int t = blockIdx.x * 256 + threadIdx.x;          // 0 .. 44*512-1
  int frag = t >> 9, rem = t & 511;
  int lane = rem >> 4, i = rem & 15;
  int g = lane >> 4, mloc = lane & 15;
  int klocal = g * 16 + i;
  int set, f;
  if      (frag < 12) { set = 0; f = frag; }
  else if (frag < 20) { set = 1; f = frag - 12; }
  else if (frag < 28) { set = 2; f = frag - 20; }
  else                { set = 3; f = frag - 28; }
  int ktile = f >> 2, mtile = f & 3;
  int m = mtile * 16 + mloc;
  int k = ktile * 32 + klocal;
  float v;
  if (set == 0) {
    // X rows: 0..63 = gathered feats (w1 cols 3..66), 64=corr(col0),
    // 65=ox(col1), 66=oy(col2), 67=fdx(col67), 68=fdy(col68), 69..95 = 0
    const float* r = w1 + (size_t)m * 69;
    if      (k < 64)  v = r[k + 3];
    else if (k == 64) v = r[0];
    else if (k == 65) v = r[1];
    else if (k == 66) v = r[2];
    else if (k == 67) v = r[67];
    else if (k == 68) v = r[68];
    else              v = 0.f;
  } else if (set == 1) v = w2[(size_t)m * 64 + k];
  else   if (set == 2) v = w3[(size_t)m * 64 + k];
  else                 v = wf[(size_t)m * 128 + k];
  wsA[t] = (_Float16)v;
}

// ---------------- Main fused kernel ----------------
static __device__ __forceinline__ void initAcc(v8f acc[4], const float* biasRow, int g) {
#pragma unroll
  for (int mt = 0; mt < 4; mt++) {
    int m0 = mt * 16 + 8 * g;
#pragma unroll
    for (int r = 0; r < 8; r++) acc[mt][r] = biasRow[m0 + r];
  }
}

// One K-tile: load the A (weight) fragment once, feed TWO N-tiles with it.
static __device__ __forceinline__ void mmK2(const _Float16* __restrict__ wsA, int fragBase,
                                            int ktile, const _Float16* row0,
                                            const _Float16* row1, int k0, int lane,
                                            v8f acc0[4], v8f acc1[4]) {
  v16h b0 = ld32h(row0 + k0);
  v16h b1 = ld32h(row1 + k0);
#pragma unroll
  for (int mt = 0; mt < 4; mt++) {
    v16h a = ld32h(wsA + (((size_t)(fragBase + ktile * 4 + mt)) << 9) + (lane << 4));
    acc0[mt] = __builtin_amdgcn_wmma_f32_16x16x32_f16(
        false, a, false, b0, (short)0, acc0[mt], false, false);
    acc1[mt] = __builtin_amdgcn_wmma_f32_16x16x32_f16(
        false, a, false, b1, (short)0, acc1[mt], false, false);
  }
}

static __device__ __forceinline__ void lreluStore(v8f acc[4], _Float16* xrow, int g) {
#pragma unroll
  for (int mt = 0; mt < 4; mt++) {
#pragma unroll
    for (int r = 0; r < 8; r++) {
      float x = acc[mt][r];
      acc[mt][r] = x > 0.f ? x : 0.1f * x;
    }
    unsigned* dst = (unsigned*)(xrow + mt * 16 + 8 * g);
#pragma unroll
    for (int r = 0; r < 8; r += 2) dst[r >> 1] = pack2(acc[mt][r], acc[mt][r + 1]);
  }
}

__global__ __launch_bounds__(128) void fuser_main(
    const _Float16* __restrict__ ptF, const float* __restrict__ ptS,
    const _Float16* __restrict__ wsA, const float* __restrict__ feat2d,
    const float* __restrict__ lf2d, const int* __restrict__ nnp,
    const float* __restrict__ b1, const float* __restrict__ b2,
    const float* __restrict__ b3, const float* __restrict__ bf,
    float* __restrict__ out) {
  __shared__ __align__(16) _Float16 Xbuf[4][32 * XSTR];   // X[k] rows 0..95, 32 pixels/wave
  __shared__ __align__(16) _Float16 F2buf[4][32 * FSTR];  // feat_2d f16
  __shared__ __align__(16) float    biasLds[4][64];

  int tid = threadIdx.x;
  int w = tid >> 5, lane = tid & 31;
  int n = lane & 15, g = lane >> 4;

  { // stage biases (all 128 threads)
    int s = tid >> 6, j = tid & 63;
    biasLds[s][j]     = (s == 0 ? b1 : b2)[j];
    biasLds[2 + s][j] = (s == 0 ? b3 : bf)[j];
  }

  size_t pix0 = (size_t)blockIdx.x * 128 + (size_t)w * 32;
  int b  = (int)(pix0 >> 16);
  int pp = (int)(pix0 & (HWPIX - 1));

  // ---- stage A for both N-tiles of this wave ----
#pragma unroll
  for (int nt = 0; nt < 2; nt++) {
    int col = nt * 16 + n;                 // pixel column within wave tile
    int p   = pp + col;                    // pixel in image (32 contiguous per wave)
    int px = p & (IMW - 1), py = p >> 8;
    int idx = nnp[b * HWPIX + p];

    _Float16* xw = &Xbuf[w][col * XSTR];
    _Float16* fw = &F2buf[w][col * FSTR];

    // gather point features (L2-resident packed array), 32 halves per lane
    const v8h* pf = (const v8h*)(ptF + (((size_t)(b * NPTS + idx)) << 6) + g * 32);
    v8h q0 = pf[0], q1 = pf[1], q2 = pf[2], q3 = pf[3];
    { // store rows g*32 .. g*32+31 of X
      v8h* xs = (v8h*)(xw + g * 32);
      xs[0] = q0; xs[1] = q1; xs[2] = q2; xs[3] = q3;
    }

    // feat_2d staging + correlation partial (channels g*32..g*32+31)
    const float* f2p = feat2d + ((size_t)(b * 64 + g * 32)) * HWPIX + p;
    unsigned* fdst = (unsigned*)(fw + g * 32);
    float corrp = 0.f;
#pragma unroll
    for (int j = 0; j < 32; j += 2) {
      float a0 = f2p[(size_t)j * HWPIX];
      float a1 = f2p[(size_t)(j + 1) * HWPIX];
      _Float16 h0, h1;
      if      (j < 8)  { h0 = q0[j & 7]; h1 = q0[(j & 7) + 1]; }
      else if (j < 16) { h0 = q1[j & 7]; h1 = q1[(j & 7) + 1]; }
      else if (j < 24) { h0 = q2[j & 7]; h1 = q2[(j & 7) + 1]; }
      else             { h0 = q3[j & 7]; h1 = q3[(j & 7) + 1]; }
      corrp += a0 * (float)h0 + a1 * (float)h1;
      fdst[j >> 1] = pack2(a0, a1);
    }
    float corr = (corrp + __shfl_xor(corrp, 16, 32)) * (1.f / 64.f);

    // special channels (f32): offset + flow delta
    float4 sv = *(const float4*)(ptS + (((size_t)(b * NPTS + idx)) << 2));
    float ox = sv.x - (float)px;
    float oy = sv.y - (float)py;
    float fdx = sv.z - lf2d[(size_t)b * 2 * HWPIX + p];
    float fdy = sv.w - lf2d[(size_t)b * 2 * HWPIX + HWPIX + p];
    if (g == 0) { // rows 64..71: corr, ox, oy, fdx, fdy, 0, 0, 0
      uint4 sp;
      sp.x = pack2(corr, ox);
      sp.y = pack2(oy, fdx);
      sp.z = pack2(fdy, 0.f);
      sp.w = 0u;
      *(uint4*)(xw + 64) = sp;
    } else {      // rows 72..95: zero pad
      uint4 z = make_uint4(0, 0, 0, 0);
      *(uint4*)(xw + 72) = z;
      *(uint4*)(xw + 80) = z;
      *(uint4*)(xw + 88) = z;
    }
  }
  __syncthreads();

  _Float16* xr0 = &Xbuf[w][n * XSTR];
  _Float16* xr1 = &Xbuf[w][(16 + n) * XSTR];
  _Float16* fr0 = &F2buf[w][n * FSTR];
  _Float16* fr1 = &F2buf[w][(16 + n) * FSTR];

  v8f acc0[4], acc1[4];
  // ---- layer 1: K = 96 ----
  initAcc(acc0, biasLds[0], g);
  initAcc(acc1, biasLds[0], g);
#pragma unroll
  for (int kt = 0; kt < 3; kt++)
    mmK2(wsA, 0, kt, xr0, xr1, kt * 32 + g * 16, lane, acc0, acc1);
  lreluStore(acc0, xr0, g);
  lreluStore(acc1, xr1, g);
  __syncthreads();
  // ---- layer 2: K = 64 ----
  initAcc(acc0, biasLds[1], g);
  initAcc(acc1, biasLds[1], g);
#pragma unroll
  for (int kt = 0; kt < 2; kt++)
    mmK2(wsA, 12, kt, xr0, xr1, kt * 32 + g * 16, lane, acc0, acc1);
  lreluStore(acc0, xr0, g);
  lreluStore(acc1, xr1, g);
  __syncthreads();
  // ---- layer 3: K = 64 ----
  initAcc(acc0, biasLds[2], g);
  initAcc(acc1, biasLds[2], g);
#pragma unroll
  for (int kt = 0; kt < 2; kt++)
    mmK2(wsA, 20, kt, xr0, xr1, kt * 32 + g * 16, lane, acc0, acc1);
  lreluStore(acc0, xr0, g);
  lreluStore(acc1, xr1, g);
  __syncthreads();
  // ---- final layer: K = 128 (x3 || feat_2d) ----
  initAcc(acc0, biasLds[3], g);
  initAcc(acc1, biasLds[3], g);
  mmK2(wsA, 28, 0, xr0, xr1, 0  + g * 16, lane, acc0, acc1);
  mmK2(wsA, 28, 1, xr0, xr1, 32 + g * 16, lane, acc0, acc1);
  mmK2(wsA, 28, 2, fr0, fr1, 0  + g * 16, lane, acc0, acc1);
  mmK2(wsA, 28, 3, fr0, fr1, 32 + g * 16, lane, acc0, acc1);

#pragma unroll
  for (int nt = 0; nt < 2; nt++) {
    v8f* acc = nt ? acc1 : acc0;
    float* op = out + ((size_t)b * 64) * HWPIX + pp + nt * 16 + n;
#pragma unroll
    for (int mt = 0; mt < 4; mt++) {
#pragma unroll
      for (int r = 0; r < 8; r++) {
        float x = acc[mt][r];
        x = x > 0.f ? x : 0.1f * x;
        op[(size_t)(mt * 16 + 8 * g + r) * HWPIX] = x;
      }
    }
  }
}

extern "C" void kernel_launch(void* const* d_in, const int* in_sizes, int n_in,
                              void* d_out, int out_size, void* d_ws, size_t ws_size,
                              hipStream_t stream) {
  const float* xy     = (const float*)d_in[0];
  const float* feat2d = (const float*)d_in[1];
  const float* feat3d = (const float*)d_in[2];
  const float* lf2d   = (const float*)d_in[3];
  const float* fl3    = (const float*)d_in[4];
  const int*   nnp    = (const int*)d_in[5];
  const float* w1 = (const float*)d_in[6];
  const float* b1 = (const float*)d_in[7];
  const float* w2 = (const float*)d_in[8];
  const float* b2 = (const float*)d_in[9];
  const float* w3 = (const float*)d_in[10];
  const float* b3 = (const float*)d_in[11];
  const float* wf = (const float*)d_in[12];
  const float* bf = (const float*)d_in[13];
  float* out = (float*)d_out;

  char* ws = (char*)d_ws;
  const size_t PTF_BYTES = (size_t)BATCH * NPTS * 64 * 2;  // 8,388,608
  const size_t PTS_BYTES = (size_t)BATCH * NPTS * 4 * 4;   // 1,048,576
  _Float16* ptF = (_Float16*)ws;
  float*    ptS = (float*)(ws + PTF_BYTES);
  _Float16* wsA = (_Float16*)(ws + PTF_BYTES + PTS_BYTES); // 45,056 bytes

  pack_points<<<(BATCH * NPTS) / 256, 256, 0, stream>>>(feat3d, xy, fl3, ptF, ptS);
  pack_weights<<<(44 * 512) / 256, 256, 0, stream>>>(w1, w2, w3, wf, wsA);
  fuser_main<<<(BATCH * HWPIX) / 128, 128, 0, stream>>>(
      ptF, ptS, wsA, feat2d, lf2d, nnp, b1, b2, b3, bf, out);
}
